// Agent_58617713656175
// MI455X (gfx1250) — matrix operand order repaired and checked by reading, not compiled, not run on hardware
//
#include <hip/hip_runtime.h>
#include <hip/hip_bf16.h>
#include <math.h>

typedef __attribute__((ext_vector_type(16))) _Float16 v16h;
typedef __attribute__((ext_vector_type(8)))  float    v8f;

__device__ __forceinline__ v8f wmma16(v16h a, v16h b, v8f c) {
  return __builtin_amdgcn_wmma_f32_16x16x32_f16(false, a, false, b, (short)0, c, false, false);
}
__device__ __forceinline__ float sigmoidf_(float x) { return 1.f / (1.f + __expf(-x)); }

#if defined(__has_builtin)
# if __has_builtin(__builtin_amdgcn_global_load_async_to_lds_b32)
#  define HAVE_ASYNC_COPY 1
# endif
#endif
#ifndef HAVE_ASYNC_COPY
# define HAVE_ASYNC_COPY 0
#endif

typedef __attribute__((address_space(1))) int gas_int;   // global
typedef __attribute__((address_space(3))) int las_int;   // LDS

__device__ __forceinline__ void wait_async0() {
#if defined(__has_builtin)
# if __has_builtin(__builtin_amdgcn_s_wait_asynccnt)
  __builtin_amdgcn_s_wait_asynccnt(0);
# else
  asm volatile("s_wait_asynccnt 0x0" ::: "memory");
# endif
#else
  asm volatile("s_wait_asynccnt 0x0" ::: "memory");
#endif
}

// ---------------- float -> half weight conversion ----------------
__global__ void __launch_bounds__(256) f2h_kernel(const float* __restrict__ s,
                                                  _Float16* __restrict__ d, int n) {
  int i = blockIdx.x * 256 + threadIdx.x;
  if (i < n) d[i] = (_Float16)s[i];
}

// ---------------- implicit-GEMM conv: one wave = 16 pixels x COUT channels ----------------
template <int CIN, int HIN, int WIN, int KH, int KW, int STRIDE,
          int COUT, int HOUT, int WOUT, bool INF, bool OUTF>
__global__ void __launch_bounds__(32) conv_wmma_t(
    const void* __restrict__ inp, const _Float16* __restrict__ w,
    const float* __restrict__ bias, void* __restrict__ outp) {
  constexpr int KHW  = KH * KW;
  constexpr int KTOT = CIN * KHW;      // 256 / 512 / 576 : multiples of 32
  constexpr int P    = HOUT * WOUT;
  constexpr int MT   = (P + 15) >> 4;
  constexpr int NACC = COUT / 16;

  int blk   = blockIdx.x;
  int img   = blk / MT;
  int mtile = blk - img * MT;

  int lane = threadIdx.x;
  int l16  = lane & 15;
  int hi   = lane >> 4;

  int pix    = mtile * 16 + l16;
  bool valid = pix < P;
  int oh = pix / WOUT, ow = pix - oh * WOUT;
  long ibase = ((long)img * CIN * HIN + (long)oh * STRIDE) * WIN + ow * STRIDE;

  const float*    inF = (const float*)inp;
  const _Float16* inH = (const _Float16*)inp;

  v8f acc[NACC] = {};
  for (int kc = 0; kc < KTOT; kc += 32) {
    v16h a;
#pragma unroll
    for (int e = 0; e < 16; ++e) {
      int k = kc + ((e >> 3) << 4) + (hi << 3) + (e & 7);
      float v = 0.f;
      if (valid) {
        int cin = k / KHW; int r = k - cin * KHW;   // compile-time divisors
        int kh  = r / KW;  int kw = r - kh * KW;
        long idx = ibase + ((long)cin * HIN + kh) * WIN + kw;
        v = INF ? inF[idx] : (float)inH[idx];
      }
      a[e] = (_Float16)v;
    }
    int kb = kc + (hi << 3);
#pragma unroll
    for (int j = 0; j < NACC; ++j) {
      int oc = j * 16 + l16;
      v16h b;
#pragma unroll
      for (int e = 0; e < 16; ++e) b[e] = w[oc * KTOT + kb + ((e >> 3) << 4) + (e & 7)];
      acc[j] = wmma16(a, b, acc[j]);
    }
  }

#pragma unroll
  for (int j = 0; j < NACC; ++j) {
    int oc = j * 16 + l16;
    float bv = bias[oc];
#pragma unroll
    for (int r = 0; r < 8; ++r) {
      int p2 = mtile * 16 + r + (hi << 3);
      if (p2 < P) {
        float v = fmaxf(acc[j][r] + bv, 0.f);       // all convs have ReLU
        long oidx = ((long)img * COUT + oc) * P + p2;
        if (OUTF) ((float*)outp)[oidx] = v;
        else      ((_Float16*)outp)[oidx] = (_Float16)v;
      }
    }
  }
}

// ---------------- generic GEMM: one wave = 16 rows x 64 cols, A reused 4x ----------------
__global__ void __launch_bounds__(32) gemm_wmma_kernel(
    const _Float16* __restrict__ A, const _Float16* __restrict__ Bw,
    const float* __restrict__ bias, const float* __restrict__ resid,
    int Ntot, int K, int relu, int outHalf, int permuteFeat, int ldo,
    void* __restrict__ outp) {
  int ngrp  = Ntot >> 6;
  int mtile = blockIdx.x / ngrp;
  int g     = blockIdx.x - mtile * ngrp;
  int n0    = g << 6;

  int lane = threadIdx.x;
  int l16  = lane & 15;
  int hi   = lane >> 4;
  long row = (long)mtile * 16 + l16;

  v8f acc[4] = {};
  for (int kc = 0; kc < K; kc += 32) {
    v16h a;
#pragma unroll
    for (int e = 0; e < 16; ++e)
      a[e] = A[row * K + kc + ((e >> 3) << 4) + (hi << 3) + (e & 7)];
    if (kc + 32 < K) {                              // gfx1250 global_prefetch_b8
      __builtin_prefetch(A + row * K + kc + 32, 0, 3);
      __builtin_prefetch(Bw + (long)(n0 + l16) * K + kc + 32, 0, 3);
    }
#pragma unroll
    for (int j = 0; j < 4; ++j) {
      int n = n0 + j * 16 + l16;
      v16h b;
#pragma unroll
      for (int e = 0; e < 16; ++e) b[e] = Bw[(long)n * K + kc + ((e >> 3) << 4) + (hi << 3) + (e & 7)];
      acc[j] = wmma16(a, b, acc[j]);
    }
  }

  float*    outF = (float*)outp;
  _Float16* outH = (_Float16*)outp;
#pragma unroll
  for (int j = 0; j < 4; ++j) {
    int n = n0 + j * 16 + l16;
    float bv = bias ? bias[n] : 0.f;
#pragma unroll
    for (int r = 0; r < 8; ++r) {
      long orow = (long)mtile * 16 + r + (hi << 3);
      float v = acc[j][r] + bv;
      if (resid) v += resid[orow * Ntot + n];
      if (relu)  v = fmaxf(v, 0.f);
      long oidx;
      if (permuteFeat) {                 // token row -> feat layout img*2304 + c*36 + p
        long img = orow / 36; long p = orow - img * 36;
        oidx = img * 2304 + (long)n * 36 + p;
      } else {
        oidx = orow * ldo + n;
      }
      if (outHalf) outH[oidx] = (_Float16)v; else outF[oidx] = v;
    }
  }
}

// ---------------- CBAM channel attention ----------------
__global__ void __launch_bounds__(64) chatt_kernel(
    const float* __restrict__ h0, const float* __restrict__ fc1w,
    const float* __restrict__ fc2w, float* __restrict__ chatt) {
  __shared__ float vmean[64], vmax[64], hmean[16], hmax[16];
  int img = blockIdx.x, c = threadIdx.x;
  const float* hp = h0 + ((long)img * 64 + c) * 36;
  float s = 0.f, m = -1e30f;
  for (int p = 0; p < 36; ++p) { float x = hp[p]; s += x; m = fmaxf(m, x); }
  vmean[c] = s * (1.f / 36.f); vmax[c] = m;
  __syncthreads();
  if (c < 16) {
    float a = 0.f;
    for (int j = 0; j < 64; ++j) a += vmean[j] * fc1w[c * 64 + j];
    hmean[c] = fmaxf(a, 0.f);
  } else if (c < 32) {
    int i = c - 16; float a = 0.f;
    for (int j = 0; j < 64; ++j) a += vmax[j] * fc1w[i * 64 + j];
    hmax[i] = fmaxf(a, 0.f);
  }
  __syncthreads();
  float o = 0.f;
  for (int i = 0; i < 16; ++i) o += (hmean[i] + hmax[i]) * fc2w[c * 16 + i];
  chatt[(long)img * 64 + c] = sigmoidf_(o);
}

// ---------------- apply channel att, spatial att, emit tokens ----------------
__global__ void __launch_bounds__(64) spatt_tok_kernel(
    const float* __restrict__ h0, const float* __restrict__ chatt,
    const float* __restrict__ spw, _Float16* __restrict__ tokh,
    float* __restrict__ tokf) {
  __shared__ float hsl[64 * 36];
  __shared__ float spm[36], spx[36], satt[36];
  int img = blockIdx.x, c = threadIdx.x;
  float ca = chatt[(long)img * 64 + c];
  const float* hp = h0 + ((long)img * 64 + c) * 36;
  for (int p = 0; p < 36; ++p) hsl[c * 36 + p] = hp[p] * ca;
  __syncthreads();
  if (c < 36) {
    float s = 0.f, m = -1e30f;
    for (int ch = 0; ch < 64; ++ch) { float x = hsl[ch * 36 + c]; s += x; m = fmaxf(m, x); }
    spm[c] = s * (1.f / 64.f); spx[c] = m;
  }
  __syncthreads();
  if (c < 36) {
    int oh = c / 6, ow = c - oh * 6;
    float a = 0.f;
    for (int kh = 0; kh < 7; ++kh) {
      int ih = oh + kh - 3; if (ih < 0 || ih >= 6) continue;
      for (int kw = 0; kw < 7; ++kw) {
        int iw = ow + kw - 3; if (iw < 0 || iw >= 6) continue;
        int p = ih * 6 + iw;
        a += spm[p] * spw[kh * 7 + kw] + spx[p] * spw[49 + kh * 7 + kw];
      }
    }
    satt[c] = sigmoidf_(a);
  }
  __syncthreads();
  for (int p = 0; p < 36; ++p) {
    float v = hsl[c * 36 + p] * satt[p];
    long ti = ((long)img * 36 + p) * 64 + c;
    tokh[ti] = (_Float16)v;
    tokf[ti] = v;
  }
}

// ---------------- per-(image,head) softmax attention over 36 tokens ----------------
__global__ void __launch_bounds__(64) attn_kernel(const _Float16* __restrict__ qkv,
                                                  _Float16* __restrict__ attnout) {
  __shared__ _Float16 ks[576], vs[576];
  int img = blockIdx.x >> 2, head = blockIdx.x & 3;
  int tid = threadIdx.x;
  const _Float16* base = qkv + (long)img * 36 * 192 + head * 16;
#if HAVE_ASYNC_COPY
  // async global->LDS staging of K and V tiles (36x16 halves each), 32-bit words
  for (int i = tid; i < 288; i += 64) {
    int t = i >> 3, dp = i & 7;
    const _Float16* gk = base + t * 192 + 64  + dp * 2;
    const _Float16* gv = base + t * 192 + 128 + dp * 2;
    __builtin_amdgcn_global_load_async_to_lds_b32(
        (gas_int*)gk, (las_int*)&ks[i * 2], 0, 0);
    __builtin_amdgcn_global_load_async_to_lds_b32(
        (gas_int*)gv, (las_int*)&vs[i * 2], 0, 0);
  }
  wait_async0();
  __syncthreads();
#else
  for (int i = tid; i < 576; i += 64) {
    int t = i >> 4, d = i & 15;
    ks[i] = base[t * 192 + 64 + d];
    vs[i] = base[t * 192 + 128 + d];
  }
  __syncthreads();
#endif
  if (tid < 36) {
    float q[16];
#pragma unroll
    for (int d = 0; d < 16; ++d) q[d] = (float)base[tid * 192 + d];
    float sc[36]; float mx = -1e30f;
    for (int u = 0; u < 36; ++u) {
      float s = 0.f;
#pragma unroll
      for (int d = 0; d < 16; ++d) s += q[d] * (float)ks[u * 16 + d];
      s *= 0.25f; sc[u] = s; mx = fmaxf(mx, s);
    }
    float sum = 0.f;
    for (int u = 0; u < 36; ++u) { float e = __expf(sc[u] - mx); sc[u] = e; sum += e; }
    float inv = 1.f / sum;
    for (int d = 0; d < 16; ++d) {
      float o = 0.f;
      for (int u = 0; u < 36; ++u) o += sc[u] * (float)vs[u * 16 + d];
      attnout[((long)img * 36 + tid) * 64 + head * 16 + d] = (_Float16)(o * inv);
    }
  }
}

// ---------------- persistent masked-GRU scan (T=64, B=32, H=128) ----------------
__global__ void __launch_bounds__(256) gru_kernel(
    const float* __restrict__ gx, const float* __restrict__ done,
    const float* __restrict__ h0in, const _Float16* __restrict__ whhh,
    const float* __restrict__ bhh, float* __restrict__ hsout) {
  __shared__ float hcur[32 * 128];
  __shared__ float hmask[32 * 128];
  __shared__ float gh[32 * 384];
  int tid = threadIdx.x;
  for (int i = tid; i < 4096; i += 256) hcur[i] = h0in[i];
  __syncthreads();

  int lane = tid & 31, wave = tid >> 5;
  int l16 = lane & 15, hi = lane >> 4;
  int mtile = wave & 1;            // this wave's 16 batch rows
  int nb    = (wave >> 1) * 6;     // this wave's 6 column tiles (96 cols)

  for (int t = 0; t < 64; ++t) {
    for (int i = tid; i < 4096; i += 256) {
      int b = i >> 7;
      hmask[i] = hcur[i] * (1.f - done[t * 32 + b]);
    }
    __syncthreads();
    // gh = hmask(32x128) @ whh^T(128x384) + bhh : A fragment reused across 6 B tiles
    {
      v8f acc[6] = {};
      for (int kc = 0; kc < 128; kc += 32) {
        v16h a;
#pragma unroll
        for (int e = 0; e < 16; ++e)
          a[e] = (_Float16)hmask[(mtile * 16 + l16) * 128 + kc +
                                 ((e >> 3) << 4) + (hi << 3) + (e & 7)];
#pragma unroll
        for (int j = 0; j < 6; ++j) {
          int n = (nb + j) * 16 + l16;
          v16h b;
#pragma unroll
          for (int e = 0; e < 16; ++e) b[e] = whhh[n * 128 + kc + ((e >> 3) << 4) + (hi << 3) + (e & 7)];
          acc[j] = wmma16(a, b, acc[j]);
        }
      }
#pragma unroll
      for (int j = 0; j < 6; ++j) {
        int n = (nb + j) * 16 + l16;
#pragma unroll
        for (int r = 0; r < 8; ++r) {
          int mr = mtile * 16 + r + (hi << 3);
          gh[mr * 384 + n] = acc[j][r] + bhh[n];
        }
      }
    }
    __syncthreads();
    for (int i = tid; i < 4096; i += 256) {
      int b = i >> 7, j = i & 127;
      const float* g = gx + (long)(t * 32 + b) * 384;
      float rg = sigmoidf_(g[j]       + gh[b * 384 + j]);
      float zg = sigmoidf_(g[128 + j] + gh[b * 384 + 128 + j]);
      float ng = tanhf(g[256 + j] + rg * gh[b * 384 + 256 + j]);
      float hv = (1.f - zg) * ng + zg * hmask[i];
      hcur[i] = hv;
      hsout[(long)(t * 32 + b) * 128 + j] = hv;
    }
    __syncthreads();
  }
}

// ---------------- critic head ----------------
__global__ void __launch_bounds__(128) critic_kernel(
    const float* __restrict__ hs, const float* __restrict__ crw,
    const float* __restrict__ crb, float* __restrict__ out) {
  __shared__ float red[128];
  int r = blockIdx.x, j = threadIdx.x;
  red[j] = hs[(long)r * 128 + j] * crw[j];
  __syncthreads();
  for (int s = 64; s > 0; s >>= 1) {
    if (j < s) red[j] += red[j + s];
    __syncthreads();
  }
  if (j == 0) out[r] = red[0] + crb[0];
}

// ================= host =================
extern "C" void kernel_launch(void* const* d_in, const int* in_sizes, int n_in,
                              void* d_out, int out_size, void* d_ws, size_t ws_size,
                              hipStream_t stream) {
  (void)in_sizes; (void)n_in; (void)out_size; (void)ws_size;
  const float* x    = (const float*)d_in[0];
  const float* done = (const float*)d_in[1];
  const float* gru0 = (const float*)d_in[2];
  const float* c1w = (const float*)d_in[3];  const float* c1b = (const float*)d_in[4];
  const float* c2w = (const float*)d_in[5];  const float* c2b = (const float*)d_in[6];
  const float* c3w = (const float*)d_in[7];  const float* c3b = (const float*)d_in[8];
  const float* fc1w = (const float*)d_in[9]; const float* fc2w = (const float*)d_in[10];
  const float* spw  = (const float*)d_in[11];
  const float* qkvw = (const float*)d_in[12]; const float* projw = (const float*)d_in[13];
  const float* projb = (const float*)d_in[14];
  const float* fcw = (const float*)d_in[15]; const float* fcb = (const float*)d_in[16];
  const float* wih = (const float*)d_in[17]; const float* whh = (const float*)d_in[18];
  const float* bih = (const float*)d_in[19]; const float* bhh = (const float*)d_in[20];
  const float* crw = (const float*)d_in[21]; const float* crb = (const float*)d_in[22];
  float* out = (float*)d_out;

  char* ws = (char*)d_ws;
  size_t off = 0;
  auto take = [&](size_t bytes) -> void* {
    off = (off + 255) & ~(size_t)255;
    void* p = ws + off; off += bytes; return p;
  };
  _Float16* w1h   = (_Float16*)take(8192 * 2);
  _Float16* w2h   = (_Float16*)take(32768 * 2);
  _Float16* w3h   = (_Float16*)take(36864 * 2);
  _Float16* wqkvh = (_Float16*)take(12288 * 2);
  _Float16* wprjh = (_Float16*)take(4096 * 2);
  _Float16* wfch  = (_Float16*)take(589824 * 2);
  _Float16* wihh  = (_Float16*)take(98304 * 2);
  _Float16* whhh  = (_Float16*)take(49152 * 2);
  _Float16* conv1o = (_Float16*)take((size_t)2048 * 32 * 784 * 2);   // 102.8 MB
  _Float16* conv2o = (_Float16*)take((size_t)2048 * 64 * 169 * 2);   // 44.3 MB
  // everything after conv2 reuses the conv1o region (dead by then); fits in 102.8 MB
  size_t offG = (size_t)((char*)conv1o - ws);
  auto takeG = [&](size_t bytes) -> void* {
    offG = (offG + 255) & ~(size_t)255;
    void* p = ws + offG; offG += bytes; return p;
  };
  float*    h0      = (float*)takeG((size_t)2048 * 64 * 36 * 4);
  float*    chatt   = (float*)takeG((size_t)2048 * 64 * 4);
  _Float16* tokh    = (_Float16*)takeG((size_t)2048 * 36 * 64 * 2);
  float*    tokf    = (float*)takeG((size_t)2048 * 36 * 64 * 4);
  _Float16* qkvb    = (_Float16*)takeG((size_t)2048 * 36 * 192 * 2);
  _Float16* attnout = (_Float16*)takeG((size_t)2048 * 36 * 64 * 2);
  _Float16* featin  = (_Float16*)takeG((size_t)2048 * 2304 * 2);
  _Float16* feat    = (_Float16*)takeG((size_t)2048 * 256 * 2);
  float*    gx      = (float*)takeG((size_t)2048 * 384 * 4);
  float*    hsout   = (float*)takeG((size_t)2048 * 128 * 4);

  auto f2h = [&](const float* s, _Float16* d, int n) {
    f2h_kernel<<<(n + 255) / 256, 256, 0, stream>>>(s, d, n);
  };
  f2h(c1w, w1h, 8192);     f2h(c2w, w2h, 32768);  f2h(c3w, w3h, 36864);
  f2h(qkvw, wqkvh, 12288); f2h(projw, wprjh, 4096);
  f2h(fcw, wfch, 589824);  f2h(wih, wihh, 98304); f2h(whh, whhh, 49152);

  // conv1: (2048,4,116,116) -> (2048,32,28,28), K=256
  conv_wmma_t<4, 116, 116, 8, 8, 4, 32, 28, 28, true, false>
      <<<2048 * 49, 32, 0, stream>>>(x, w1h, c1b, conv1o);
  // conv2: -> (2048,64,13,13), K=512
  conv_wmma_t<32, 28, 28, 4, 4, 2, 64, 13, 13, false, false>
      <<<2048 * 11, 32, 0, stream>>>(conv1o, w2h, c2b, conv2o);
  // conv3: -> (2048,64,6,6) float, K=576
  conv_wmma_t<64, 13, 13, 3, 3, 2, 64, 6, 6, false, true>
      <<<2048 * 3, 32, 0, stream>>>(conv2o, w3h, c3b, h0);

  chatt_kernel<<<2048, 64, 0, stream>>>(h0, fc1w, fc2w, chatt);
  spatt_tok_kernel<<<2048, 64, 0, stream>>>(h0, chatt, spw, tokh, tokf);

  // QKV: (73728,64) @ (64,192)
  gemm_wmma_kernel<<<4608 * 3, 32, 0, stream>>>(
      tokh, wqkvh, nullptr, nullptr, 192, 64, 0, 1, 0, 192, qkvb);
  attn_kernel<<<2048 * 4, 64, 0, stream>>>(qkvb, attnout);
  // proj + residual -> featin (permuted to c*36+p layout)
  gemm_wmma_kernel<<<4608 * 1, 32, 0, stream>>>(
      attnout, wprjh, projb, tokf, 64, 64, 0, 1, 1, 64, featin);
  // FC 2304 -> 256, ReLU
  gemm_wmma_kernel<<<128 * 4, 32, 0, stream>>>(
      featin, wfch, fcb, nullptr, 256, 2304, 1, 1, 0, 256, feat);
  // gx = feat @ wih^T + bih : (2048,384) float
  gemm_wmma_kernel<<<128 * 6, 32, 0, stream>>>(
      feat, wihh, bih, nullptr, 384, 256, 0, 0, 0, 384, gx);
  // sequential GRU scan (single persistent workgroup, WMMA from LDS)
  gru_kernel<<<1, 256, 0, stream>>>(gx, done, gru0, whhh, bhh, hsout);
  critic_kernel<<<2048, 128, 0, stream>>>(hsout, crw, crb, out);
}
